// MultiHeadCausalSelfAttention_11948599017550
// MI455X (gfx1250) — matrix operand order repaired and checked by reading, not compiled, not run on hardware
//
#include <hip/hip_runtime.h>
#include <hip/hip_bf16.h>

#define D_EMB 1024
#define HEADS 16
#define HD    64
#define BATCH 4
#define SEQ   2048
#define M_TOT (BATCH * SEQ)   // 8192
#define N_QKV (3 * D_EMB)     // 3072
#define NKIT  (D_EMB / 32)    // 32 K-steps

typedef __attribute__((ext_vector_type(16))) __bf16        v16bf;
typedef __attribute__((ext_vector_type(8)))  float         v8f;
typedef __attribute__((ext_vector_type(8)))  unsigned short us8;

union FragAB {
    v16bf v;
    us8   h[2];
    unsigned short e[16];
};

__device__ __forceinline__ unsigned short f2bf(float f) {
    union { float f; unsigned int u; } x; x.f = f;
    unsigned int u = x.u;
    if (((u >> 23) & 0xFFu) == 0xFFu) return (unsigned short)(u >> 16); // Inf/NaN
    return (unsigned short)((u + 0x7FFFu + ((u >> 16) & 1u)) >> 16);    // RNE
}

__device__ __forceinline__ v8f zero8() {
    v8f z;
#pragma unroll
    for (int i = 0; i < 8; ++i) z[i] = 0.0f;
    return z;
}

// gfx1250 async copy: global -> LDS, 16B per lane, tracked by ASYNCcnt.
// LDS byte offset = low 32 bits of the generic pointer to __shared__ memory.
__device__ __forceinline__ void async_g2l_b128(const void* gptr, void* lptr) {
    unsigned loff = (unsigned)(unsigned long long)lptr;
    asm volatile("global_load_async_to_lds_b128 %0, %1, off"
                 :: "v"(loff), "v"(gptr) : "memory");
}
template <int N>
__device__ __forceinline__ void wait_async() {
    asm volatile("s_wait_asynccnt %0" :: "i"(N) : "memory");
}

// ---------------------------------------------------------------- converts
__global__ void cvt_f32_bf16(const float* __restrict__ src,
                             unsigned short* __restrict__ dst, int n) {
    for (int i = blockIdx.x * blockDim.x + threadIdx.x; i < n;
         i += gridDim.x * blockDim.x)
        dst[i] = f2bf(src[i]);
}

// dst[k][colOff + n] = src[n][k]  (f32 [N,K] -> bf16 transposed [K, dstStride])
__global__ void cvt_transpose(const float* __restrict__ src,
                              unsigned short* __restrict__ dst,
                              int N, int K, int dstStride, int colOff) {
    for (int i = blockIdx.x * blockDim.x + threadIdx.x; i < N * K;
         i += gridDim.x * blockDim.x) {
        const int n = i / K, k = i - n * K;
        dst[(size_t)k * dstStride + colOff + n] = f2bf(src[i]);
    }
}

// ---------------------------------------------------------------- QKV GEMM
// C[m, n] = sum_k xb[m,k] * wqkvT[k,n]   (M=8192, N=3072, K=1024)
// 256 threads = 8 waves (4 M x 2 N), block tile 128x128, wave tile 32x64.
// A tile double-buffered in LDS via async copies; fully unrolled K loop.
__global__ __launch_bounds__(256)
void qkv_gemm(const unsigned short* __restrict__ xb,      // [M, D] bf16
              const unsigned short* __restrict__ wqkvT,   // [D, 3D] bf16
              unsigned short* __restrict__ Qh,            // [B*H, T, HD] (x1/8)
              unsigned short* __restrict__ Kt,            // [B*H, HD, T]
              unsigned short* __restrict__ Vh) {          // [B*H, T, HD]
    __shared__ unsigned short lA[2][128][32];
    const int tid  = threadIdx.x;
    const int wave = tid >> 5;
    const int lane = tid & 31;
    const int hh   = lane >> 4;
    const int r    = lane & 15;
    const int blockM = blockIdx.y * 128;
    const int blockN = blockIdx.x * 128;
    const int waveM  = (wave & 3) * 32;
    const int waveN  = (wave >> 2) * 64;

    v8f acc[2][4];
#pragma unroll
    for (int mi = 0; mi < 2; ++mi)
#pragma unroll
        for (int ni = 0; ni < 4; ++ni) acc[mi][ni] = zero8();

    const int arow = tid >> 1;              // A-stage: 32B per thread
    const int acol = (tid & 1) * 16;
    const unsigned short* aSrc = &xb[(size_t)(blockM + arow) * D_EMB + acol];
    const unsigned short* bSrc = &wqkvT[blockN + waveN];

    auto step = [&](int cur, int kb) {      // one K-step: 2 A frags, 4 B, 8 wmma
        FragAB a[2];
#pragma unroll
        for (int mi = 0; mi < 2; ++mi) {
            a[mi].h[0] = *(const us8*)&lA[cur][waveM + mi * 16 + r][hh * 8];
            a[mi].h[1] = *(const us8*)&lA[cur][waveM + mi * 16 + r][16 + hh * 8];
        }
#pragma unroll
        for (int ni = 0; ni < 4; ++ni) {
            FragAB b;
#pragma unroll
            for (int i = 0; i < 16; ++i)
                b.e[i] = bSrc[(size_t)(kb + (i < 8 ? hh * 8 + i : 8 + hh * 8 + i)) * N_QKV + ni * 16 + r];
#pragma unroll
            for (int mi = 0; mi < 2; ++mi)
                acc[mi][ni] = __builtin_amdgcn_wmma_f32_16x16x32_bf16(
                    false, a[mi].v, false, b.v, (short)0, acc[mi][ni],
                    false, false);
        }
    };

    // prologue: kick off tile 0
    async_g2l_b128(aSrc,     &lA[0][arow][acol]);
    async_g2l_b128(aSrc + 8, &lA[0][arow][acol + 8]);

#pragma unroll
    for (int kt = 0; kt < NKIT - 1; ++kt) { // always prefetches: branch-free
        const int kb  = kt * 32;
        const int cur = kt & 1;
        const unsigned short* s = aSrc + (size_t)(kb + 32);
        async_g2l_b128(s,     &lA[cur ^ 1][arow][acol]);
        async_g2l_b128(s + 8, &lA[cur ^ 1][arow][acol + 8]);
        wait_async<2>();                    // oldest 2 (current tile) done
        __syncthreads();
        __builtin_prefetch(&wqkvT[(size_t)(kb + 32 + hh * 16 + r) * N_QKV +
                                  blockN + waveN], 0, 3);
        step(cur, kb);
        __syncthreads();                    // done reading cur before reuse
    }
    wait_async<0>();                        // peeled last K-step
    __syncthreads();
    step((NKIT - 1) & 1, (NKIT - 1) * 32);

    // epilogue: Q (pre-scaled) -> [B*H,T,HD]; K -> [B*H,HD,T]; V -> [B*H,T,HD]
    const int which = blockN >> 10;         // uniform per block: 0=Q 1=K 2=V
#pragma unroll
    for (int mi = 0; mi < 2; ++mi)
#pragma unroll
        for (int ni = 0; ni < 4; ++ni)
#pragma unroll
            for (int v = 0; v < 8; ++v) {
                const int m = blockM + waveM + mi * 16 + v + 8 * hh;
                const int nn = (blockN + waveN + ni * 16 + r) & 1023;
                const int head = nn >> 6;
                const int hd = nn & 63;
                const int bb = m >> 11;     // m / SEQ
                const int t = m & 2047;
                const size_t bhh = (size_t)(bb * HEADS + head);
                const float val = acc[mi][ni][v];
                if (which == 0)
                    Qh[(bhh * SEQ + t) * HD + hd] = f2bf(val * 0.125f);
                else if (which == 1)
                    Kt[(bhh * HD + hd) * SEQ + t] = f2bf(val);
                else
                    Vh[(bhh * SEQ + t) * HD + hd] = f2bf(val);
            }
}

// ---------------------------------------------------------------- attention
// One wave per 16-query tile, flash-style streaming over 32-key blocks.
__global__ __launch_bounds__(128)
void attn(const unsigned short* __restrict__ Q,    // [B*H, T, HD]
          const unsigned short* __restrict__ Kt,   // [B*H, HD, T]
          const unsigned short* __restrict__ V,    // [B*H, T, HD]
          unsigned short* __restrict__ ctx) {      // [B, T, D] bf16
    __shared__ unsigned short pLds[4][16][32];     // per-wave P transpose bounce
    const int wave = threadIdx.x >> 5;
    const int lane = threadIdx.x & 31;
    const int hh = lane >> 4, r = lane & 15;
    const int bh = blockIdx.y;
    const int b = bh / HEADS, head = bh % HEADS;
    const int qbase = blockIdx.x * 64 + wave * 16;
    const unsigned short* Qp = Q  + (size_t)bh * SEQ * HD;
    const unsigned short* Kp = Kt + (size_t)bh * HD * SEQ;
    const unsigned short* Vp = V  + (size_t)bh * SEQ * HD;

    // Q fragments (A layout), loaded once: two 16x32 chunks over Hd=64
    FragAB aq[2];
#pragma unroll
    for (int c = 0; c < 2; ++c) {
        const size_t base = (size_t)(qbase + r) * HD + c * 32;
        aq[c].h[0] = *(const us8*)&Qp[base + hh * 8];
        aq[c].h[1] = *(const us8*)&Qp[base + 16 + hh * 8];
    }

    v8f o[4];
#pragma unroll
    for (int n = 0; n < 4; ++n) o[n] = zero8();
    float mrow[8], lrow[8];
#pragma unroll
    for (int v = 0; v < 8; ++v) { mrow[v] = -__builtin_inff(); lrow[v] = 0.0f; }

    const int nj = (qbase + 47) >> 5;              // causal key-block count
    for (int j = 0; j < nj; ++j) {
        const int kbase = j * 32;

        // S = Q K^T : Kt rows are head-dims, so B frags are strided gathers
        v8f st[2];
#pragma unroll
        for (int t = 0; t < 2; ++t) {
            st[t] = zero8();
#pragma unroll
            for (int c = 0; c < 2; ++c) {
                FragAB bk;
#pragma unroll
                for (int i = 0; i < 16; ++i)
                    bk.e[i] = Kp[(size_t)(c * 32 + (i < 8 ? hh * 8 + i : 8 + hh * 8 + i)) * SEQ + kbase + t * 16 + r];
                st[t] = __builtin_amdgcn_wmma_f32_16x16x32_bf16(
                    false, aq[c].v, false, bk.v, (short)0, st[t], false, false);
            }
        }

        // causal mask + online softmax (rows live in 16-lane halves)
        float p0v[8], p1v[8];
#pragma unroll
        for (int v = 0; v < 8; ++v) {
            const int q = qbase + v + 8 * hh;
            float s0 = (kbase + r      <= q) ? st[0][v] : -__builtin_inff();
            float s1 = (kbase + 16 + r <= q) ? st[1][v] : -__builtin_inff();
            float mx = fmaxf(s0, s1);
#pragma unroll
            for (int d = 1; d < 16; d <<= 1) mx = fmaxf(mx, __shfl_xor(mx, d, 32));
            const float mnew  = fmaxf(mrow[v], mx);
            const float alpha = __expf(mrow[v] - mnew);
            const float p0 = __expf(s0 - mnew);
            const float p1 = __expf(s1 - mnew);
            float rs = p0 + p1;
#pragma unroll
            for (int d = 1; d < 16; d <<= 1) rs += __shfl_xor(rs, d, 32);
            lrow[v] = lrow[v] * alpha + rs;
            mrow[v] = mnew;
            p0v[v] = p0; p1v[v] = p1;
#pragma unroll
            for (int n = 0; n < 4; ++n) o[n][v] *= alpha;
        }

        // transpose P (C layout -> A layout) through wave-private LDS
#pragma unroll
        for (int v = 0; v < 8; ++v) {
            pLds[wave][v + 8 * hh][r]      = f2bf(p0v[v]);
            pLds[wave][v + 8 * hh][16 + r] = f2bf(p1v[v]);
        }
        asm volatile("s_wait_dscnt 0" ::: "memory");   // wave-local RAW fence
        __builtin_amdgcn_wave_barrier();
        FragAB ap;
        ap.h[0] = *(const us8*)&pLds[wave][r][hh * 8];
        ap.h[1] = *(const us8*)&pLds[wave][r][16 + hh * 8];

        // O += P V   (B fragments gathered down V columns)
#pragma unroll
        for (int n = 0; n < 4; ++n) {
            FragAB bv;
#pragma unroll
            for (int i = 0; i < 16; ++i)
                bv.e[i] = Vp[(size_t)(kbase + (i < 8 ? hh * 8 + i : 8 + hh * 8 + i)) * HD + n * 16 + r];
            o[n] = __builtin_amdgcn_wmma_f32_16x16x32_bf16(
                false, ap.v, false, bv.v, (short)0, o[n], false, false);
        }
    }

    // normalize and merge heads into [B, T, D]
#pragma unroll
    for (int v = 0; v < 8; ++v) {
        const float inv = 1.0f / lrow[v];
        const int q = qbase + v + 8 * hh;
#pragma unroll
        for (int n = 0; n < 4; ++n) {
            const size_t dst =
                ((size_t)(b * SEQ + q)) * D_EMB + head * HD + n * 16 + r;
            ctx[dst] = f2bf(o[n][v] * inv);
        }
    }
}

// ---------------------------------------------------------------- out proj
__global__ __launch_bounds__(256)
void oproj_gemm(const unsigned short* __restrict__ ctxb,  // [M, D] bf16
                const unsigned short* __restrict__ woT,   // [D, D] bf16 (transposed)
                const float* __restrict__ bo,
                float* __restrict__ out) {                // [M, D] f32
    __shared__ unsigned short lA[2][128][32];
    const int tid  = threadIdx.x;
    const int wave = tid >> 5;
    const int lane = tid & 31;
    const int hh   = lane >> 4;
    const int r    = lane & 15;
    const int blockM = blockIdx.y * 128;
    const int blockN = blockIdx.x * 128;
    const int waveM  = (wave & 3) * 32;
    const int waveN  = (wave >> 2) * 64;

    v8f acc[2][4];
#pragma unroll
    for (int mi = 0; mi < 2; ++mi)
#pragma unroll
        for (int ni = 0; ni < 4; ++ni) acc[mi][ni] = zero8();

    const int arow = tid >> 1;
    const int acol = (tid & 1) * 16;
    const unsigned short* aSrc = &ctxb[(size_t)(blockM + arow) * D_EMB + acol];
    const unsigned short* bSrc = &woT[blockN + waveN];

    auto step = [&](int cur, int kb) {
        FragAB a[2];
#pragma unroll
        for (int mi = 0; mi < 2; ++mi) {
            a[mi].h[0] = *(const us8*)&lA[cur][waveM + mi * 16 + r][hh * 8];
            a[mi].h[1] = *(const us8*)&lA[cur][waveM + mi * 16 + r][16 + hh * 8];
        }
#pragma unroll
        for (int ni = 0; ni < 4; ++ni) {
            FragAB b;
#pragma unroll
            for (int i = 0; i < 16; ++i)
                b.e[i] = bSrc[(size_t)(kb + (i < 8 ? hh * 8 + i : 8 + hh * 8 + i)) * D_EMB + ni * 16 + r];
#pragma unroll
            for (int mi = 0; mi < 2; ++mi)
                acc[mi][ni] = __builtin_amdgcn_wmma_f32_16x16x32_bf16(
                    false, a[mi].v, false, b.v, (short)0, acc[mi][ni],
                    false, false);
        }
    };

    async_g2l_b128(aSrc,     &lA[0][arow][acol]);
    async_g2l_b128(aSrc + 8, &lA[0][arow][acol + 8]);

#pragma unroll
    for (int kt = 0; kt < NKIT - 1; ++kt) {
        const int kb  = kt * 32;
        const int cur = kt & 1;
        const unsigned short* s = aSrc + (size_t)(kb + 32);
        async_g2l_b128(s,     &lA[cur ^ 1][arow][acol]);
        async_g2l_b128(s + 8, &lA[cur ^ 1][arow][acol + 8]);
        wait_async<2>();
        __syncthreads();
        __builtin_prefetch(&woT[(size_t)(kb + 32 + hh * 16 + r) * D_EMB +
                                blockN + waveN], 0, 3);
        step(cur, kb);
        __syncthreads();
    }
    wait_async<0>();
    __syncthreads();
    step((NKIT - 1) & 1, (NKIT - 1) * 32);

#pragma unroll
    for (int mi = 0; mi < 2; ++mi)
#pragma unroll
        for (int ni = 0; ni < 4; ++ni)
#pragma unroll
            for (int v = 0; v < 8; ++v) {
                const int m = blockM + waveM + mi * 16 + v + 8 * hh;
                const int n = blockN + waveN + ni * 16 + r;
                out[(size_t)m * D_EMB + n] = acc[mi][ni][v] + bo[n];
            }
}

// ---------------------------------------------------------------- launch
extern "C" void kernel_launch(void* const* d_in, const int* in_sizes, int n_in,
                              void* d_out, int out_size, void* d_ws,
                              size_t ws_size, hipStream_t stream) {
    const float* x  = (const float*)d_in[0];
    const float* Wq = (const float*)d_in[1];
    const float* Wk = (const float*)d_in[2];
    const float* Wv = (const float*)d_in[3];
    const float* Wo = (const float*)d_in[4];
    const float* bo = (const float*)d_in[5];

    char* ws = (char*)d_ws;
    unsigned short* xb    = (unsigned short*)(ws);                       // 16 MB
    unsigned short* wqkvT = (unsigned short*)(ws + ((size_t)16 << 20));  //  6 MB
    unsigned short* woT   = (unsigned short*)(ws + ((size_t)22 << 20));  //  2 MB
    unsigned short* Qh    = (unsigned short*)(ws + ((size_t)24 << 20));  // 16 MB
    unsigned short* Kt    = (unsigned short*)(ws + ((size_t)40 << 20));  // 16 MB
    unsigned short* Vh    = (unsigned short*)(ws + ((size_t)56 << 20));  // 16 MB
    unsigned short* ctxb  = (unsigned short*)(ws + ((size_t)72 << 20));  // 16 MB

    cvt_f32_bf16<<<512, 256, 0, stream>>>(x, xb, M_TOT * D_EMB);
    cvt_transpose<<<1024, 256, 0, stream>>>(Wq, wqkvT, D_EMB, D_EMB, N_QKV, 0);
    cvt_transpose<<<1024, 256, 0, stream>>>(Wk, wqkvT, D_EMB, D_EMB, N_QKV, D_EMB);
    cvt_transpose<<<1024, 256, 0, stream>>>(Wv, wqkvT, D_EMB, D_EMB, N_QKV, 2 * D_EMB);
    cvt_transpose<<<1024, 256, 0, stream>>>(Wo, woT, D_EMB, D_EMB, D_EMB, 0);

    qkv_gemm<<<dim3(N_QKV / 128, M_TOT / 128), 256, 0, stream>>>(
        xb, wqkvT, Qh, Kt, Vh);
    attn<<<dim3(SEQ / 64, BATCH * HEADS), 128, 0, stream>>>(Qh, Kt, Vh, ctxb);
    oproj_gemm<<<dim3(D_EMB / 128, M_TOT / 128), 256, 0, stream>>>(
        ctxb, woT, bo, (float*)d_out);
}
